// MultiHeadAttentionLayer_12644383719677
// MI455X (gfx1250) — matrix operand order repaired and checked by reading, not compiled, not run on hardware
//
#include <hip/hip_runtime.h>
#include <hip/hip_bf16.h>

typedef __attribute__((ext_vector_type(16))) __bf16 v16bf;
typedef __attribute__((ext_vector_type(8)))  float  v8f;

#define NNODES 50000
#define NEDGES 600000
#define DIN    128
#define DMODEL 128          // N_HEADS * D_HEAD
#define QKVCOLS 384

// ---- workspace layout (bytes) ----
// [0, 98304)                      : fragQKV  (24 ct * 4 kb * 32 lanes * 16 bf16)
// [98304, 131072)                 : fragWE   ( 8 ct * 4 kb * 32 lanes * 16 bf16)
// [131072, 131072+76800000)       : qkv f32  [N][384]  (Q|K|V)
// then wV f32 [N][128], then z f32 [N][8]  (contiguous, zeroed each call)
#define OFF_FRAGWE  98304
#define OFF_QKV     131072
#define OFF_WV      (131072 + 76800000)
#define OFF_Z       (131072 + 76800000 + 25600000)

__device__ __forceinline__ float clip5(float x) { return fminf(fmaxf(x, -5.f), 5.f); }

// ------------------------------------------------------------------ zero
__global__ void zero_f32(float* __restrict__ p, int n) {
    int i = blockIdx.x * 256 + threadIdx.x;
    if (i < n) p[i] = 0.0f;
}

// ------------------------------------------------------------ pack weights
// B-fragment packing in this code's contiguous k order (32x16 K-block): lane = khalf*16 + n,
// VGPR v holds K pair (khalf*16 + 2v, +1) for column n.  frag element e = 2v+p.
__global__ void pack_weights(const float* __restrict__ WQ, const float* __restrict__ WK,
                             const float* __restrict__ WV, const float* __restrict__ WE,
                             __bf16* __restrict__ fragQKV, __bf16* __restrict__ fragWE) {
    int idx = blockIdx.x * 256 + threadIdx.x;
    if (idx < 24 * 4 * 32 * 16) {
        int e = idx & 15, lane = (idx >> 4) & 31, kb = (idx >> 9) & 3, ct = idx >> 11;
        int n = lane & 15, khalf = lane >> 4;
        int v = e >> 1, p = e & 1;
        int k = kb * 32 + khalf * 16 + v * 2 + p;
        int ng = ct * 16 + n;
        float w = (ng < 128) ? WQ[k * 128 + ng]
                : (ng < 256) ? WK[k * 128 + (ng - 128)]
                             : WV[k * 128 + (ng - 256)];
        fragQKV[idx] = (__bf16)w;
    }
    int idx2 = idx - 24 * 4 * 32 * 16;
    if (idx2 >= 0 && idx2 < 8 * 4 * 32 * 16) {
        int e = idx2 & 15, lane = (idx2 >> 4) & 31, kb = (idx2 >> 9) & 3, ct = idx2 >> 11;
        int n = lane & 15, khalf = lane >> 4;
        int v = e >> 1, p = e & 1;
        int k = kb * 32 + khalf * 16 + v * 2 + p;
        fragWE[idx2] = (__bf16)WE[k * 128 + ct * 16 + n];
    }
}

// A-fragment build (16x32, contiguous k order matching the B packing): lane = half*16 + m,
// VGPR v: k = kb*32 + half*16 + v*2.
__device__ __forceinline__ void build_a_frags(const float* __restrict__ rows, // [16][128] in LDS
                                              int m, int half, v16bf a[4]) {
#pragma unroll
    for (int kb = 0; kb < 4; ++kb) {
#pragma unroll
        for (int v = 0; v < 8; ++v) {
            int k = kb * 32 + half * 16 + v * 2;
            a[kb][2 * v]     = (__bf16)rows[m * 128 + k];
            a[kb][2 * v + 1] = (__bf16)rows[m * 128 + k + 1];
        }
    }
}

// ------------------------------------------------------------- QKV GEMM
// grid = (3125 row-tiles, 3 col-groups), block = 256 (8 waves).
// wave w computes the 16x16 tile at col-tile ct = blockIdx.y*8 + w.
__global__ __launch_bounds__(256) void qkv_gemm(const float* __restrict__ x,
                                                const __bf16* __restrict__ fragQKV,
                                                float* __restrict__ qkv) {
    __shared__ float xs[16 * 128];
    int tid = threadIdx.x;
    int row0 = blockIdx.x * 16;
    {
        const float4* xg = (const float4*)(x + (size_t)row0 * 128);
        float4* s4 = (float4*)xs;
#pragma unroll
        for (int j = 0; j < 2; ++j) s4[tid + j * 256] = xg[tid + j * 256];
    }
    __syncthreads();

    int wave = tid >> 5, lane = tid & 31;
    int ct = blockIdx.y * 8 + wave;           // 0..23
    int m = lane & 15, half = lane >> 4;

    v16bf a[4];
    build_a_frags(xs, m, half, a);

    v8f c = {};
#pragma unroll
    for (int kb = 0; kb < 4; ++kb) {
        v16bf b = *(const v16bf*)(fragQKV + ((size_t)(ct * 4 + kb) * 32 + lane) * 16);
        c = __builtin_amdgcn_wmma_f32_16x16x32_bf16(false, a[kb], false, b,
                                                    (short)0, c, false, false);
    }
    int n = lane & 15;
#pragma unroll
    for (int r = 0; r < 8; ++r) {
        int mm = r + half * 8;                // C layout: M = r + 8*half, N = lane&15
        qkv[(size_t)(row0 + mm) * QKVCOLS + ct * 16 + n] = c[r];
    }
}

// --------------------------------------------------------- fused edge kernel
// Each wave owns a 16-edge tile; per head h (== WMMA N-tile, D_HEAD=16):
//   e_proj tile via WMMA(edge_attr_bf16, WE_frag), then per C-row r:
//   alpha = clip(k[row]*q[col]/4)*e_proj -> e_out, butterfly-sum -> alphax,
//   scatter wV += v[row]*alphax and z += alphax with hardware f32 atomics.
__global__ __launch_bounds__(256) void edge_attn(const float* __restrict__ edge_attr,
                                                 const int* __restrict__ e_row,
                                                 const int* __restrict__ e_col,
                                                 const __bf16* __restrict__ fragWE,
                                                 const float* __restrict__ qkv,
                                                 float* __restrict__ e_out,
                                                 float* __restrict__ wV,
                                                 float* __restrict__ z) {
    __shared__ float es[8 * 16 * 128];        // 64 KB, 8 KB per wave
    int tid = threadIdx.x, wave = tid >> 5, lane = tid & 31;
    int e0 = (blockIdx.x * 8 + wave) * 16;
    float* my = es + wave * 2048;

#pragma unroll 4
    for (int rr = 0; rr < 16; ++rr) {         // stage 16 edge rows (coalesced)
        int ee = e0 + rr; ee = (ee < NEDGES) ? ee : (NEDGES - 1);
        *(float4*)(my + rr * 128 + lane * 4) =
            *(const float4*)(edge_attr + (size_t)ee * 128 + lane * 4);
    }
    __syncthreads();
    if (e0 >= NEDGES) return;                 // wave-uniform; EXEC all-ones below

    int m = lane & 15, half = lane >> 4, n = m;
    v16bf a[4];
    build_a_frags(my, m, half, a);

    int rowi[8], coli[8];
#pragma unroll
    for (int r = 0; r < 8; ++r) {
        int ee = e0 + half * 8 + r; ee = (ee < NEDGES) ? ee : (NEDGES - 1);
        rowi[r] = e_row[ee];
        coli[r] = e_col[ee];
    }

#pragma unroll 1
    for (int h = 0; h < 8; ++h) {
        v8f c = {};
#pragma unroll
        for (int kb = 0; kb < 4; ++kb) {
            v16bf b = *(const v16bf*)(fragWE + ((size_t)(h * 4 + kb) * 32 + lane) * 16);
            c = __builtin_amdgcn_wmma_f32_16x16x32_bf16(false, a[kb], false, b,
                                                        (short)0, c, false, false);
        }
#pragma unroll
        for (int r = 0; r < 8; ++r) {
            int ee = e0 + half * 8 + r;
            bool valid = (ee < NEDGES);
            // gathers hit L2: whole qkv table (76.8 MB) resides in the 192 MB L2
            float kg = qkv[(size_t)rowi[r] * QKVCOLS + 128 + h * 16 + n];
            float qg = qkv[(size_t)coli[r] * QKVCOLS +       h * 16 + n];
            float al = clip5(kg * qg * 0.25f) * c[r];
            if (valid) e_out[(size_t)ee * 128 + h * 16 + n] = al;
            float s = al;                      // sum over D_HEAD within 16-lane half
            s += __shfl_xor(s, 1);
            s += __shfl_xor(s, 2);
            s += __shfl_xor(s, 4);
            s += __shfl_xor(s, 8);
            float ax = __expf(clip5(s));
            float vg = qkv[(size_t)rowi[r] * QKVCOLS + 256 + h * 16 + n];
            if (valid) unsafeAtomicAdd(&wV[(size_t)coli[r] * 128 + h * 16 + n], vg * ax);
            if (valid && n == 0) unsafeAtomicAdd(&z[(size_t)coli[r] * 8 + h], ax);
        }
    }
}

// ---------------------------------------------------------------- finalize
__global__ void finalize(const float* __restrict__ wV, const float* __restrict__ z,
                         float* __restrict__ h_out) {
    int i = blockIdx.x * 256 + threadIdx.x;
    if (i < NNODES * DMODEL) {
        int node = i >> 7, h = (i >> 4) & 7;
        h_out[i] = wV[i] / (z[node * 8 + h] + 1e-6f);
    }
}

extern "C" void kernel_launch(void* const* d_in, const int* in_sizes, int n_in,
                              void* d_out, int out_size, void* d_ws, size_t ws_size,
                              hipStream_t stream) {
    const float* x         = (const float*)d_in[0];
    const float* edge_attr = (const float*)d_in[1];
    const int*   edge_index= (const int*)d_in[2];   // [2][E]: row then col
    const float* WQ        = (const float*)d_in[3];
    const float* WK        = (const float*)d_in[4];
    const float* WV        = (const float*)d_in[5];
    const float* WE        = (const float*)d_in[6];

    char* ws = (char*)d_ws;
    __bf16* fragQKV = (__bf16*)ws;
    __bf16* fragWE  = (__bf16*)(ws + OFF_FRAGWE);
    float*  qkv     = (float*)(ws + OFF_QKV);
    float*  wV      = (float*)(ws + OFF_WV);
    float*  zb      = (float*)(ws + OFF_Z);

    float* h_out = (float*)d_out;
    float* e_out = h_out + (size_t)NNODES * DMODEL;

    // wV and z are contiguous -> one zeroing pass (must re-zero every call)
    int zc = NNODES * DMODEL + NNODES * 8;
    zero_f32<<<(zc + 255) / 256, 256, 0, stream>>>(wV, zc);
    pack_weights<<<65536 / 256, 256, 0, stream>>>(WQ, WK, WV, WE, fragQKV, fragWE);
    qkv_gemm<<<dim3(NNODES / 16, 3), 256, 0, stream>>>(x, fragQKV, qkv);
    edge_attn<<<(NEDGES / 16 + 7) / 8, 256, 0, stream>>>(edge_attr, edge_index,
                                                         edge_index + NEDGES,
                                                         fragWE, qkv, e_out, wV, zb);
    finalize<<<(NNODES * DMODEL + 255) / 256, 256, 0, stream>>>(wV, zb, h_out);
}
